// FourierKANLayer_14293651161795
// MI455X (gfx1250) — matrix operand order repaired and checked by reading, not compiled, not run on hardware
//
#include <hip/hip_runtime.h>
#include <cstdint>
#include <cstddef>

// ---------------------------------------------------------------------------
// FourierKAN  ==  Y[2048,512] = T[2048,32768] @ W[32768,512] + bias
//   T[b,(s,i,g)] = {cos,sin}(x[b,i]*(g+1)),  W = fourier_coeffs flattened.
// MI455X plan:
//   * stage T (128 MiB) and W (32 MiB) as f16 WMMA *fragments* in scratch ->
//     both fit the 192 MB global L2 -> GEMM operands are L2/LDS/TDM-fed.
//   * GEMM: B fragments DMA'd into LDS by the Tensor Data Mover (one
//     tensor_load_to_lds per 8-K-step chunk, double-buffered 64 KB of the
//     320 KB/WGP), synced with s_wait_tensorcnt + workgroup barrier.
//     A is per-wave coalesced global_load_b128; 32 WMMAs per chunk body.
// ---------------------------------------------------------------------------

typedef __attribute__((ext_vector_type(16))) _Float16 v16h;
typedef __attribute__((ext_vector_type(8)))  float    v8f;
typedef __attribute__((ext_vector_type(4)))  uint32_t v4u;
typedef __attribute__((ext_vector_type(8)))  uint32_t v8u;

#define BATCH   2048
#define IN_DIM  512
#define GRID_N  32
#define OUT_DIM 512
#define KDIM    (2 * IN_DIM * GRID_N)   // 32768
#define KT      (KDIM / 32)             // 1024 K-tiles (one 16x16x32 WMMA each)
#define MT      (BATCH / 16)            // 128 M-tiles
#define NTL     (OUT_DIM / 16)          // 32  N-tiles
#define FRAG_BYTES 1024                 // 32 lanes * 32 B (v16h) per fragment

#define CHUNK        8                          // K-tiles staged per barrier
#define NCHUNK       (KT / CHUNK)               // 128
#define BCHUNK_BYTES (CHUNK * 4 * FRAG_BYTES)   // 32 KB (4 N-tiles per block)

// ---------------------------------------------------------------------------
// Phase W: coeffs f32 [32768,512] -> B-matrix f16 fragments.
// B-layout (ISA 7.12.2, 16-bit B 32x16): lane = khalf*16 + n; lane holds its
// column's K = (p>=8 ? 16:0) + khalf*8 + (p&7) at position p, packed 2/VGPR.
// ---------------------------------------------------------------------------
__global__ void build_W(const float* __restrict__ coeffs,
                        _Float16* __restrict__ wfrag)
{
    const int lane  = threadIdx.x & 31;
    const int wid   = threadIdx.x >> 5;
    const int khalf = lane >> 4;
    const int lrow  = lane & 15;
    const int kt    = blockIdx.x;               // 0..1023
    const int ntile = blockIdx.y * 8 + wid;     // 0..31
    const int o     = ntile * 16 + lrow;

    v16h w;
#pragma unroll
    for (int p = 0; p < 16; ++p) {
        const size_t k = (size_t)kt * 32 + (size_t)(((p >> 3) << 4) + khalf * 8 + (p & 7));
        w[p] = (_Float16)coeffs[k * OUT_DIM + o];   // coalesced over o per p
    }
    *(v16h*)((char*)wfrag + ((size_t)ntile * KT + kt) * FRAG_BYTES + lane * 32) = w;
}

// ---------------------------------------------------------------------------
// Phase T: x f32 -> A-matrix f16 fragments; cos tile (kt=i) and sin tile
// (kt=512+i) produced from ONE sincos pass (zero trig recompute overall).
// A-layout (ISA 7.12.2, 16-bit A 16x32): lane&15 = row; lane<16 holds
// K{0..7,16..23}, lane>=16 holds K{8..15,24..31}; position p maps to
// K = (p>=8 ? 16:0) + khalf*8 + (p&7).
// ---------------------------------------------------------------------------
__global__ void build_T(const float* __restrict__ x,
                        _Float16* __restrict__ tfrag)
{
    const int lane  = threadIdx.x & 31;
    const int wid   = threadIdx.x >> 5;
    const int khalf = lane >> 4;
    const int lrow  = lane & 15;
    const int i     = blockIdx.x;               // 0..511   (input dim)
    const int mtile = blockIdx.y * 8 + wid;     // 0..127
    const int row   = mtile * 16 + lrow;

    const float xv = x[(size_t)row * IN_DIM + i];

    v16h fc, fs;
#pragma unroll
    for (int p = 0; p < 16; ++p) {
        const int K = ((p >> 3) << 4) + khalf * 8 + (p & 7);   // 0..31 in-tile
        float s, c;
        __sincosf(xv * (float)(K + 1), &s, &c);                // freqs 1..32
        fc[p] = (_Float16)c;
        fs[p] = (_Float16)s;
    }
    char* base = (char*)tfrag + (size_t)mtile * KT * FRAG_BYTES + (size_t)lane * 32;
    *(v16h*)(base + (size_t)i * FRAG_BYTES)            = fc;   // kt = i       (cos)
    *(v16h*)(base + (size_t)(IN_DIM + i) * FRAG_BYTES) = fs;   // kt = 512 + i (sin)
}

// ---------------------------------------------------------------------------
// Phase GEMM: grid (16,8) = 128 blocks x 8 waves.  Wave = one 16-row M-tile
// x 64 cols (4 N-tiles).  B fragments DMA'd chunk-wise into LDS by the TDM
// (double-buffered); A direct from L2; 32 WMMAs per straight-line chunk body.
// ---------------------------------------------------------------------------
__global__ void __launch_bounds__(256)
fkan_gemm(const _Float16* __restrict__ tfrag,
          const _Float16* __restrict__ wfrag,
          const float* __restrict__ bias,
          float* __restrict__ out)
{
    __shared__ alignas(128) char smem[2 * BCHUNK_BYTES];   // 64 KB of 320 KB/WGP

    const int tid   = threadIdx.x;
    const int lane  = tid & 31;
    const int wid   = tid >> 5;
    const int khalf = lane >> 4;
    const int lrow  = lane & 15;
    const int mtile = blockIdx.x * 8 + wid;     // grid.x = 16 -> 0..127
    const int ntb   = blockIdx.y * 4;           // grid.y = 8  -> n-tiles ntb..ntb+3

    const char* aptr  = (const char*)tfrag + (size_t)mtile * KT * FRAG_BYTES
                        + (size_t)lane * 32;
    const char* wbase = (const char*)wfrag;

    v8f acc[4];
#pragma unroll
    for (int nt = 0; nt < 4; ++nt)
        acc[nt] = v8f{0.f, 0.f, 0.f, 0.f, 0.f, 0.f, 0.f, 0.f};

    // ---- TDM descriptor (ISA 8.3/8.4).  2D tile: 4 rows (N-tiles) x 8192 B
    // (8 K-tiles of fragments), row stride = KT*FRAG_BYTES = 1 MiB.
    // Units: data_size = 3 (8 bytes) -> row = 1024 units, stride = 131072.
    const uint32_t lds_base = (uint32_t)(uintptr_t)&smem[0];

    v8u g1;
    g1[0] = 3u << 16;            // [17:16] data_size=8B; mask/pad/iterate = 0
    g1[1] = 1024u << 16;         // [63:48]  tensor_dim0 low16  (=1024 units)
    g1[2] = 4u << 16;            // [95:80..] tensor_dim0 hi=0 | tensor_dim1=4
    g1[3] = 1024u << 16;         // [127:112] tile_dim0 = 1024 units
    g1[4] = 4u;                  // [143:128] tile_dim1 = 4 ; tile_dim2 = 0 (2D)
    g1[5] = 131072u;             // [191:160] tensor_dim0_stride low32
    g1[6] = 0u;                  // stride hi | tensor_dim1_stride lo (unused)
    g1[7] = 0u;

    // prologue: DMA chunk 0 into LDS buffer 0
    if (wid == 0) {
        const uint64_t ga = (uint64_t)(uintptr_t)
            (wbase + (size_t)ntb * KT * FRAG_BYTES);           // kt = 0
        v4u g0;
        g0[0] = 1u;                                            // count=1
        g0[1] = lds_base;                                      // lds_addr
        g0[2] = (uint32_t)ga;                                  // global_addr lo
        g0[3] = (uint32_t)((ga >> 32) & 0x01FFFFFFu) | (2u << 30); // hi | type=2
        asm volatile("tensor_load_to_lds %0, %1" :: "s"(g0), "s"(g1) : "memory");
        __builtin_amdgcn_s_wait_tensorcnt(0);
    }
    __syncthreads();

    for (int ch = 0; ch < NCHUNK; ++ch) {
        const int ktbase = ch * CHUNK;

        // issue DMA for the NEXT chunk into the other buffer (its previous
        // readers were retired by the barrier that ended chunk ch-1)
        if (wid == 0 && ch + 1 < NCHUNK) {
            const uint64_t ga = (uint64_t)(uintptr_t)
                (wbase + ((size_t)ntb * KT + (ktbase + CHUNK)) * FRAG_BYTES);
            v4u g0;
            g0[0] = 1u;
            g0[1] = lds_base + (uint32_t)(((ch + 1) & 1) * BCHUNK_BYTES);
            g0[2] = (uint32_t)ga;
            g0[3] = (uint32_t)((ga >> 32) & 0x01FFFFFFu) | (2u << 30);
            asm volatile("tensor_load_to_lds %0, %1" :: "s"(g0), "s"(g1) : "memory");
        }

        // compute this chunk: straight-line 8 K-steps x 4 WMMAs.
        // TDM stores the tile contiguously row-major -> LDS layout [nt][kk].
        const char* lbase = &smem[(ch & 1) * BCHUNK_BYTES];
#pragma unroll
        for (int kk = 0; kk < CHUNK; ++kk) {
            const v16h a = *(const v16h*)(aptr + (size_t)(ktbase + kk) * FRAG_BYTES);
#pragma unroll
            for (int nt = 0; nt < 4; ++nt) {
                const v16h b = *(const v16h*)(lbase
                                + (nt * CHUNK + kk) * FRAG_BYTES + lane * 32);
                // (neg_a, A, neg_b, B, c_mod, C, reuse_a, reuse_b)
                acc[nt] = __builtin_amdgcn_wmma_f32_16x16x32_f16(
                    false, a, false, b, (short)0, acc[nt], false, false);
            }
        }

        // DMA completion (issuing wave) then block-wide rendezvous
        if (wid == 0 && ch + 1 < NCHUNK)
            __builtin_amdgcn_s_wait_tensorcnt(0);
        __syncthreads();
    }

    // Epilogue: C/D layout (ISA 7.12.2): N = lane&15, M = r + 8*khalf.
    const int rowbase = mtile * 16 + khalf * 8;
#pragma unroll
    for (int nt = 0; nt < 4; ++nt) {
        const int col = (ntb + nt) * 16 + lrow;
        const float bv = bias[col];
#pragma unroll
        for (int r = 0; r < 8; ++r)
            out[(size_t)(rowbase + r) * OUT_DIM + col] = acc[nt][r] + bv;
    }
}

// ---------------------------------------------------------------------------
extern "C" void kernel_launch(void* const* d_in, const int* in_sizes, int n_in,
                              void* d_out, int out_size, void* d_ws, size_t ws_size,
                              hipStream_t stream)
{
    const float* x      = (const float*)d_in[0];   // [2048, 512]
    const float* coeffs = (const float*)d_in[1];   // [2, 512, 32, 512]
    const float* bias   = (const float*)d_in[2];   // [1, 512]
    float*       out    = (float*)d_out;           // [2048, 512]

    char* ws = (char*)d_ws;
    const size_t wbytes = (size_t)NTL * KT * FRAG_BYTES;   //  32 MiB (W fragments)
    _Float16* wfrag = (_Float16*)ws;
    _Float16* tfrag = (_Float16*)(ws + wbytes);            // 128 MiB (T fragments)
    // total scratch: 160 MiB -> resident in MI455X's 192 MB global L2

    build_W<<<dim3(KT, NTL / 8), 256, 0, stream>>>(coeffs, wfrag);
    build_T<<<dim3(IN_DIM, MT / 8), 256, 0, stream>>>(x, tfrag);
    fkan_gemm<<<dim3(MT / 8, NTL / 4), 256, 0, stream>>>(tfrag, wfrag, bias, out);

    (void)in_sizes; (void)n_in; (void)out_size; (void)ws_size;
}
